// AR_26645977104796
// MI455X (gfx1250) — matrix operand order repaired and checked by reading, not compiled, not run on hardware
//
#include <hip/hip_runtime.h>
#include <hip/hip_bf16.h>
#include <math.h>

// ---------------------------------------------------------------------------
// Types for CDNA5 WMMA / TDM
// ---------------------------------------------------------------------------
typedef __attribute__((ext_vector_type(16))) _Float16 v16h;
typedef __attribute__((ext_vector_type(8)))  float    v8f;
typedef __attribute__((ext_vector_type(4)))  unsigned u32x4;
typedef __attribute__((ext_vector_type(8)))  int      i32x8;
typedef __attribute__((ext_vector_type(4)))  int      i32x4;

union F16Frag { v16h v; unsigned u[8]; };

// ---------------------------------------------------------------------------
// Tensor Data Mover: 2-D f32 tile  global -> LDS   (ISA 08_async_tensor §8)
//   tile_d0 elems per row, tile_d1 rows, row stride tens_stride0 elems.
//   pad_on: +1 dword of LDS padding after every 64 dwords (rows of 64 f32
//   land on a 65-float LDS pitch).
// Issue from ONE wave only (TDM ignores EXEC); caller must __syncthreads().
// ---------------------------------------------------------------------------
__device__ inline void tdm_load_f32(const float* gsrc, float* ldst,
                                    unsigned tile_d0, unsigned tile_d1,
                                    unsigned tens_stride0, int pad_on)
{
    unsigned long long ga = (unsigned long long)(size_t)gsrc;
    unsigned lds = (unsigned)(size_t)ldst;       // flat LDS addr low 32b = LDS offset

    u32x4 g0;
    g0.x = 1u;                                   // count=1, user descriptor
    g0.y = lds;                                  // D#.lds_addr
    g0.z = (unsigned)ga;                         // D#.global_addr[31:0]
    g0.w = (unsigned)(ga >> 32) | (2u << 30);    // global_addr[56:32] | type=2

    unsigned w0 = (2u << 16);                    // data_size = 4 bytes
    if (pad_on) w0 |= (1u << 20) | (5u << 22);   // pad_enable, interval=64dw, amount=1dw
    i32x8 g1;
    g1[0] = (int)w0;
    g1[1] = (int)(tile_d0 << 16);                // tensor_dim0[15:0]
    g1[2] = (int)(tile_d1 << 16);                // tensor_dim1[15:0]
    g1[3] = (int)(tile_d0 << 16);                // tile_dim0
    g1[4] = (int)tile_d1;                        // tile_dim1 (tile_dim2 = 0)
    g1[5] = (int)tens_stride0;                   // tensor_dim0_stride[31:0]
    g1[6] = 0;
    g1[7] = 0;

    i32x4 z4 = {0, 0, 0, 0};
#if defined(__clang_major__) && (__clang_major__ >= 23)
    i32x8 z8 = {0, 0, 0, 0, 0, 0, 0, 0};
    __builtin_amdgcn_tensor_load_to_lds(g0, g1, z4, z4, z8, 0);
#else
    __builtin_amdgcn_tensor_load_to_lds(g0, g1, z4, z4, 0);
#endif
}

// db12 (sym12-like) filter
__constant__ float c_reclo[12] = {
    0.11154074335008017f,  0.4946238903983854f,   0.7511339080215775f,
    0.3152503517092432f,  -0.22626469396516913f, -0.12976686756709563f,
    0.09750160558707936f,  0.02752286553001629f, -0.031582039318031156f,
    0.0005538422009938016f, 0.004777257511010651f, -0.00107730108499558f };

// ---------------------------------------------------------------------------
// 1. Wavelet denoise: one block per row of 2048
// ---------------------------------------------------------------------------
__device__ float select_abs_kth(const float* a, int n, int k, int* redI, int tid)
{
    unsigned lo = 0u, hi = 0x7f800000u;
    while (lo < hi) {
        unsigned mid = (lo + hi) >> 1;
        int cnt = 0;
        for (int i = tid; i < n; i += 256)
            if (__float_as_uint(fabsf(a[i])) <= mid) cnt++;
        redI[tid] = cnt; __syncthreads();
        for (int s2 = 128; s2 > 0; s2 >>= 1) {
            if (tid < s2) redI[tid] += redI[tid + s2];
            __syncthreads();
        }
        int total = redI[0]; __syncthreads();
        if (total >= k + 1) hi = mid; else lo = mid + 1;
    }
    return __uint_as_float(lo);
}

__global__ __launch_bounds__(256) void wavelet_kernel(const float* __restrict__ xn,
                                                      float* __restrict__ xc)
{
    __shared__ float bufA[2048];
    __shared__ float bufB[2048];
    __shared__ float cds[2112];
    __shared__ int   redI[256];

    const int tid = threadIdx.x;
    const int row = blockIdx.x;
    const float* x = xn + (size_t)row * 2048;

    const int ls[9] = {2048, 1029, 520, 265, 138, 74, 42, 26, 18};
    const int co[8] = {0, 1029, 1549, 1814, 1952, 2026, 2068, 2094};

    // TDM: DMA the whole row into LDS (1 x 2048 f32 tile), wave 0 only
    if (tid < 32) {
        tdm_load_f32(x, bufA, 2048u, 1u, 2048u, 0);
        __builtin_amdgcn_s_wait_tensorcnt((short)0);
    }
    __syncthreads();

    // ---- forward DWT, 8 levels ----
    float* src = bufA; float* dst = bufB;
    for (int lev = 0; lev < 8; ++lev) {
        int n = ls[lev], no = ls[lev + 1];
        float* cd = cds + co[lev];
        for (int o = tid; o < no; o += 256) {
            float sa = 0.f, sd = 0.f;
            #pragma unroll
            for (int t = 0; t < 12; ++t) {
                int j = 2 * o + t - 10;
                if (j < 0)  j = -j - 1;
                if (j >= n) j = 2 * n - 1 - j;
                float v = src[j];
                sa += v * c_reclo[t];
                sd += v * (((t & 1) ? -1.f : 1.f) * c_reclo[11 - t]);
            }
            dst[o] = sa;
            cd[o]  = sd;
        }
        __syncthreads();
        float* tmp = src; src = dst; dst = tmp;
    }

    // ---- soft-threshold first 3 detail levels ----
    const float tscale = sqrtf(2.f * logf(2048.f));
    for (int lev = 0; lev < 3; ++lev) {
        int n = ls[lev + 1];
        float* cd = cds + co[lev];
        float m0 = select_abs_kth(cd, n, (n - 1) / 2, redI, tid);
        float m1 = select_abs_kth(cd, n, n / 2, redI, tid);
        float th = 0.5f * (m0 + m1) / 0.6745f * tscale;
        for (int i = tid; i < n; i += 256) {
            float v = cd[i];
            float a = fabsf(v) - th;
            cd[i] = (a > 0.f) ? ((v > 0.f ? 1.f : -1.f) * a) : 0.f;
        }
        __syncthreads();
    }

    // ---- zero final approximation, inverse DWT ----
    for (int i = tid; i < ls[8]; i += 256) src[i] = 0.f;
    __syncthreads();

    for (int lev = 7; lev >= 0; --lev) {
        int n = ls[lev + 1];
        const float* cd = cds + co[lev];
        int outl = 2 * n - 10;
        for (int i = tid; i < outl; i += 256) {
            float s = 0.f;
            #pragma unroll
            for (int t = 0; t < 12; ++t) {
                int j = i + t;
                if (j > 0) {
                    int jj = j - 1;
                    if (!(jj & 1)) {
                        int q = jj >> 1;
                        if (q < n) {
                            s += src[q] * c_reclo[11 - t];
                            s += cd[q] * (((t & 1) ? 1.f : -1.f) * c_reclo[t]);
                        }
                    }
                }
            }
            dst[i] = s;
        }
        __syncthreads();
        float* tmp = src; src = dst; dst = tmp;
    }
    for (int i = tid; i < 2048; i += 256) xc[(size_t)row * 2048 + i] = src[i];
}

// ---------------------------------------------------------------------------
// 2. Implicit-GEMM conv via WMMA.
//    Block 256 thr = 8 waves; macro-tile 128(M) x 32(N); wave-tile 32x16.
// ---------------------------------------------------------------------------
__global__ __launch_bounds__(256) void conv_wmma_kernel(
    const float* __restrict__ in, const float* __restrict__ w,
    float* __restrict__ out, int B, int Cin, int Cout, int Lin, int Lout, int up)
{
    __shared__ _Float16 As[128][40];   // [m][k] patches
    __shared__ _Float16 Bs[32][40];    // [n][k] weights

    const int tid  = threadIdx.x;
    const int lane = tid & 31;
    const int wid  = tid >> 5;
    const int wm   = (wid >> 1) << 5;      // 0,32,64,96
    const int wn   = (wid & 1) << 4;       // 0,16
    const int lh   = lane >> 4;
    const int lm   = lane & 15;
    const int m0   = blockIdx.x * 128;
    const int n0   = blockIdx.y * 32;

    const int Kreal = Cin * 5;
    const int KT    = (Kreal + 31) >> 5;

    v8f acc0 = {}, acc1 = {};

    for (int kt = 0; kt < KT; ++kt) {
        const int k0 = kt << 5;
        __syncthreads();
        // stage patch tile (128 x 32)
        #pragma unroll
        for (int it = 0; it < 16; ++it) {
            int e = tid + it * 256;
            int r = e >> 5, c = e & 31;
            int m = m0 + r;
            int b = m / Lout, p = m - b * Lout;
            int k = k0 + c;
            float v = 0.f;
            if (k < Kreal) {
                int ci = k / 5, t = k - ci * 5;
                int j = p + t - 2;
                if (j >= 0 && j < Lout) {
                    int src = up ? (j >> 1) : j;
                    v = in[((size_t)b * Cin + ci) * Lin + src];
                }
            }
            As[r][c] = (_Float16)v;
        }
        // stage weight tile (32 x 32); w is [Cout][Cin*5] row-major
        #pragma unroll
        for (int it = 0; it < 4; ++it) {
            int e = tid + it * 256;
            int r = e >> 5, c = e & 31;
            int k = k0 + c;
            float v = (k < Kreal) ? w[(size_t)(n0 + r) * Kreal + k] : 0.f;
            Bs[r][c] = (_Float16)v;
        }
        if (kt + 1 < KT)   // prefetch next weight rows (global_prefetch_b8)
            __builtin_prefetch(&w[(size_t)(n0 + (tid & 31)) * Kreal + k0 + 32], 0, 1);
        __syncthreads();

        F16Frag a0, a1, b0;
        #pragma unroll
        for (int p = 0; p < 8; ++p) {
            int ka = 2 * p + 16 * lh;
            a0.u[p] = *(const unsigned*)&As[wm + lm][ka];
            a1.u[p] = *(const unsigned*)&As[wm + 16 + lm][ka];
            int kb = 2 * p + 16 * lh;
            b0.u[p] = *(const unsigned*)&Bs[wn + lm][kb];
        }
        acc0 = __builtin_amdgcn_wmma_f32_16x16x32_f16(false, a0.v, false, b0.v, (short)0, acc0, false, false);
        acc1 = __builtin_amdgcn_wmma_f32_16x16x32_f16(false, a1.v, false, b0.v, (short)0, acc1, false, false);
    }

    // epilogue: write out[b, co, p]
    #pragma unroll
    for (int mi = 0; mi < 2; ++mi) {
        #pragma unroll
        for (int r = 0; r < 8; ++r) {
            int gm = m0 + wm + 16 * mi + r + 8 * lh;
            int gn = n0 + wn + lm;
            int b = gm / Lout, p = gm - b * Lout;
            float v = (mi == 0) ? acc0[r] : acc1[r];
            out[((size_t)b * Cout + gn) * Lout + p] = v;
        }
    }
}

// final decoder conv (Cout = 1): simple VALU kernel over repeat(x,2)
__global__ void convup_kernel(const float* __restrict__ x, const float* __restrict__ w,
                              float* __restrict__ y, int B, int Cin, int Cout, int Lin)
{
    int Lo  = 2 * Lin;
    int idx = blockIdx.x * 256 + threadIdx.x;
    int tot = B * Cout * Lo;
    if (idx >= tot) return;
    int p  = idx % Lo;
    int t2 = idx / Lo;
    int c  = t2 % Cout;
    int bb = t2 / Cout;
    float s = 0.f;
    for (int ci = 0; ci < Cin; ++ci) {
        const float* xr = x + ((size_t)bb * Cin + ci) * Lin;
        const float* wr = w + ((size_t)c * Cin + ci) * 5;
        #pragma unroll
        for (int k = 0; k < 5; ++k) {
            int j = p + k - 2;
            if (j >= 0 && j < Lo) s += xr[j >> 1] * wr[k];
        }
    }
    y[idx] = s;
}

__global__ __launch_bounds__(256) void bn_stats_kernel(const float* __restrict__ x,
                                                       float* __restrict__ st,
                                                       int B, int C, int L)
{
    __shared__ float r1[256], r2[256];
    int c = blockIdx.x, tid = threadIdx.x;
    int tot = B * L;
    float s1 = 0.f, s2 = 0.f;
    for (int i = tid; i < tot; i += 256) {
        int bb = i / L, l = i - bb * L;
        float v = x[((size_t)bb * C + c) * L + l];
        s1 += v; s2 += v * v;
    }
    r1[tid] = s1; r2[tid] = s2; __syncthreads();
    for (int s = 128; s > 0; s >>= 1) {
        if (tid < s) { r1[tid] += r1[tid + s]; r2[tid] += r2[tid + s]; }
        __syncthreads();
    }
    if (tid == 0) {
        float m = r1[0] / (float)tot;
        st[2 * c]     = m;
        st[2 * c + 1] = r2[0] / (float)tot - m * m;
    }
}

__global__ void bn_relu_pool_kernel(const float* __restrict__ x, const float* __restrict__ st,
                                    const float* __restrict__ g, const float* __restrict__ b,
                                    float* __restrict__ y, int B, int C, int L)
{
    int Lo  = L >> 1;
    int idx = blockIdx.x * 256 + threadIdx.x;
    int tot = B * C * Lo;
    if (idx >= tot) return;
    int i  = idx % Lo;
    int t2 = idx / Lo;
    int c  = t2 % C;
    int bb = t2 / C;
    float m = st[2 * c], var = st[2 * c + 1];
    float sc = g[c] * rsqrtf(var + 1e-5f);
    float sh = b[c] - m * sc;
    float best = -3e38f;
    #pragma unroll
    for (int t = 0; t < 4; ++t) {
        int j = 2 * i - 1 + t;
        if (j >= 0 && j < L) {
            float v = fmaxf(x[((size_t)bb * C + c) * L + j] * sc + sh, 0.f);
            best = fmaxf(best, v);
        }
    }
    y[((size_t)bb * C + c) * Lo + i] = best;
}

__global__ void bn_relu_kernel(const float* __restrict__ x, const float* __restrict__ st,
                               const float* __restrict__ g, const float* __restrict__ b,
                               float* __restrict__ y, int C, int L, int tot)
{
    int idx = blockIdx.x * 256 + threadIdx.x;
    if (idx >= tot) return;
    int c = (idx / L) % C;
    float m = st[2 * c], var = st[2 * c + 1];
    float sc = g[c] * rsqrtf(var + 1e-5f);
    float sh = b[c] - m * sc;
    y[idx] = fmaxf(x[idx] * sc + sh, 0.f);
}

// ---------------------------------------------------------------------------
// 3. WMMA GEMM: C[M,N] = op(A[M,K] @ W[N,K]^T + bias) (+res)
//    macro-tile 64x128, wave-tile 32x32 -> 4 v_wmma / K-step; double-buffered.
// ---------------------------------------------------------------------------
__global__ __launch_bounds__(256) void gemm_wmma_kernel(
    const float* __restrict__ A, const float* __restrict__ Wt,
    const float* __restrict__ bias, const float* __restrict__ res,
    float* __restrict__ Cm, int M, int N, int K, int relu)
{
    __shared__ _Float16 As[2][64][40];    // [buf][m][k]
    __shared__ _Float16 Bs[2][128][40];   // [buf][n][k]

    const int tid  = threadIdx.x;
    const int lane = tid & 31;
    const int wid  = tid >> 5;
    const int wm   = (wid >> 2) << 5;     // 0 / 32
    const int wn   = (wid & 3) << 5;      // 0,32,64,96
    const int lh   = lane >> 4;
    const int lm   = lane & 15;
    const int m0   = blockIdx.x * 64;
    const int n0   = blockIdx.y * 128;

    const int KT = K >> 5;

    v8f acc00 = {}, acc01 = {}, acc10 = {}, acc11 = {};

    // prologue: stage K-tile 0 directly into LDS buffer 0
    #pragma unroll
    for (int it = 0; it < 8; ++it) {
        int e = tid + it * 256;
        As[0][e >> 5][e & 31] = (_Float16)A[(size_t)(m0 + (e >> 5)) * K + (e & 31)];
    }
    #pragma unroll
    for (int it = 0; it < 16; ++it) {
        int e = tid + it * 256;
        Bs[0][e >> 5][e & 31] = (_Float16)Wt[(size_t)(n0 + (e >> 5)) * K + (e & 31)];
    }
    __syncthreads();

    for (int kt = 0; kt < KT; ++kt) {
        const int cur = kt & 1, nxt = cur ^ 1;
        const bool havenext = (kt + 1 < KT);

        float ra[8], rb[16];
        if (havenext) {
            const int k1 = (kt + 1) << 5;
            #pragma unroll
            for (int it = 0; it < 8; ++it) {
                int e = tid + it * 256;
                ra[it] = A[(size_t)(m0 + (e >> 5)) * K + k1 + (e & 31)];
            }
            #pragma unroll
            for (int it = 0; it < 16; ++it) {
                int e = tid + it * 256;
                rb[it] = Wt[(size_t)(n0 + (e >> 5)) * K + k1 + (e & 31)];
            }
        }

        F16Frag a0, a1, b0, b1;
        #pragma unroll
        for (int p = 0; p < 8; ++p) {
            int ka = 2 * p + 16 * lh;
            a0.u[p] = *(const unsigned*)&As[cur][wm + lm][ka];
            a1.u[p] = *(const unsigned*)&As[cur][wm + 16 + lm][ka];
            int kb = 2 * p + 16 * lh;
            b0.u[p] = *(const unsigned*)&Bs[cur][wn + lm][kb];
            b1.u[p] = *(const unsigned*)&Bs[cur][wn + 16 + lm][kb];
        }
        acc00 = __builtin_amdgcn_wmma_f32_16x16x32_f16(false, a0.v, false, b0.v, (short)0, acc00, false, false);
        acc01 = __builtin_amdgcn_wmma_f32_16x16x32_f16(false, a0.v, false, b1.v, (short)0, acc01, false, false);
        acc10 = __builtin_amdgcn_wmma_f32_16x16x32_f16(false, a1.v, false, b0.v, (short)0, acc10, false, false);
        acc11 = __builtin_amdgcn_wmma_f32_16x16x32_f16(false, a1.v, false, b1.v, (short)0, acc11, false, false);

        if (havenext) {
            #pragma unroll
            for (int it = 0; it < 8; ++it) {
                int e = tid + it * 256;
                As[nxt][e >> 5][e & 31] = (_Float16)ra[it];
            }
            #pragma unroll
            for (int it = 0; it < 16; ++it) {
                int e = tid + it * 256;
                Bs[nxt][e >> 5][e & 31] = (_Float16)rb[it];
            }
        }
        __syncthreads();
    }

    #pragma unroll
    for (int mi = 0; mi < 2; ++mi) {
        #pragma unroll
        for (int ni = 0; ni < 2; ++ni) {
            #pragma unroll
            for (int r = 0; r < 8; ++r) {
                int gm = m0 + wm + 16 * mi + r + 8 * lh;
                int gn = n0 + wn + 16 * ni + lm;
                float v;
                if (mi == 0) v = (ni == 0) ? acc00[r] : acc01[r];
                else         v = (ni == 0) ? acc10[r] : acc11[r];
                if (bias) v += bias[gn];
                if (relu) v = fmaxf(v, 0.f);
                if (res)  v += res[(size_t)gm * N + gn];
                Cm[(size_t)gm * N + gn] = v;
            }
        }
    }
}

// ---------------------------------------------------------------------------
// 4. Transformer helpers
// ---------------------------------------------------------------------------
__global__ void build_x0_kernel(const float* __restrict__ f, const float* __restrict__ cls,
                                float* __restrict__ X)
{
    int idx = blockIdx.x * 256 + threadIdx.x;
    const int tot = 128 * 129 * 256;
    if (idx >= tot) return;
    int d  = idx & 255;
    int t2 = idx >> 8;
    int s  = t2 % 129;
    int bb = t2 / 129;
    float base = (s == 0) ? cls[d]
                          : f[((size_t)bb * 256 + d) * 128 + (s - 1)];
    float freq = __expf(-9.210340371976184f * (float)(d & ~1) / 256.f);
    float pe   = (d & 1) ? __cosf((float)s * freq) : __sinf((float)s * freq);
    X[idx] = base + pe;
}

__global__ __launch_bounds__(256) void attention_kernel(const float* __restrict__ qkv,
                                                        float* __restrict__ ctx)
{
    __shared__ float Ks[129][65];
    __shared__ float qs[64];
    __shared__ float sc[132];
    __shared__ float red[256];

    const int tid = threadIdx.x;
    const int b   = blockIdx.x >> 2;
    const int h   = blockIdx.x & 3;
    const size_t rowbase = (size_t)b * 129;

    // TDM: DMA K block (129 rows x 64 f32, row stride 768) into Ks with
    // 1-dword padding per 64 dwords -> 65-float LDS pitch. Wave 0 issues.
    if (tid < 32) {
        tdm_load_f32(qkv + rowbase * 768 + 256 + (size_t)h * 64,
                     &Ks[0][0], 64u, 129u, 768u, 1);
        __builtin_amdgcn_s_wait_tensorcnt((short)0);
    }
    __syncthreads();

    for (int s = 0; s < 129; ++s) {
        if (tid < 64) qs[tid] = qkv[(rowbase + s) * 768 + h * 64 + tid];
        __syncthreads();

        float myv = -3e38f;
        if (tid < 129 && tid <= s) {
            float a = 0.f;
            #pragma unroll 8
            for (int d = 0; d < 64; ++d) a += qs[d] * Ks[tid][d];
            myv = a * 0.125f;   // 1/sqrt(64)
        }
        red[tid] = myv; __syncthreads();
        for (int s2 = 128; s2 > 0; s2 >>= 1) {
            if (tid < s2) red[tid] = fmaxf(red[tid], red[tid + s2]);
            __syncthreads();
        }
        float mx = red[0]; __syncthreads();

        float p = (tid < 129 && tid <= s) ? __expf(myv - mx) : 0.f;
        red[tid] = p; __syncthreads();
        for (int s2 = 128; s2 > 0; s2 >>= 1) {
            if (tid < s2) red[tid] += red[tid + s2];
            __syncthreads();
        }
        float denom = red[0]; __syncthreads();
        if (tid < 129) sc[tid] = p / denom;
        __syncthreads();

        if (tid < 64) {
            float a = 0.f;
            for (int j = 0; j <= s; ++j)
                a += sc[j] * qkv[(rowbase + j) * 768 + 512 + h * 64 + tid];
            ctx[(rowbase + s) * 256 + h * 64 + tid] = a;
        }
        __syncthreads();
    }
}

__global__ __launch_bounds__(256) void layernorm_kernel(const float* __restrict__ x,
                                                        const float* __restrict__ g,
                                                        const float* __restrict__ b,
                                                        float* __restrict__ y)
{
    __shared__ float red[256];
    int row = blockIdx.x, tid = threadIdx.x;
    float v = x[(size_t)row * 256 + tid];
    red[tid] = v; __syncthreads();
    for (int s = 128; s > 0; s >>= 1) { if (tid < s) red[tid] += red[tid + s]; __syncthreads(); }
    float m = red[0] / 256.f; __syncthreads();
    float d = v - m;
    red[tid] = d * d; __syncthreads();
    for (int s = 128; s > 0; s >>= 1) { if (tid < s) red[tid] += red[tid + s]; __syncthreads(); }
    float var = red[0] / 256.f; __syncthreads();
    y[(size_t)row * 256 + tid] = d * rsqrtf(var + 1e-5f) * g[tid] + b[tid];
}

// ---------------------------------------------------------------------------
// 5. Losses
// ---------------------------------------------------------------------------
__global__ void zero_kernel(float* p, int n)
{
    int i = blockIdx.x * 256 + threadIdx.x;
    if (i < n) p[i] = 0.f;
}

__global__ __launch_bounds__(256) void sqdiff_kernel(const float* __restrict__ a,
                                                     const float* __restrict__ b,
                                                     float* __restrict__ acc, int n)
{
    __shared__ float red[256];
    int tid = threadIdx.x;
    float s = 0.f;
    for (int i = blockIdx.x * 256 + tid; i < n; i += gridDim.x * 256) {
        float d = a[i] - b[i];
        s += d * d;
    }
    red[tid] = s; __syncthreads();
    for (int s2 = 128; s2 > 0; s2 >>= 1) { if (tid < s2) red[tid] += red[tid + s2]; __syncthreads(); }
    if (tid == 0) atomicAdd(acc, red[0]);
}

__global__ __launch_bounds__(256) void ar_loss_kernel(const float* __restrict__ X,
                                                      const float* __restrict__ f,
                                                      float* __restrict__ acc)
{
    __shared__ float red[256];
    const int n = 128 * 128 * 256;
    int tid = threadIdx.x;
    float s = 0.f;
    for (int i = blockIdx.x * 256 + tid; i < n; i += gridDim.x * 256) {
        int d  = i & 255;
        int t2 = i >> 8;
        int t  = t2 % 128;
        int bb = t2 / 128;
        float xv = X[((size_t)(bb * 129 + t + 1)) * 256 + d];
        float fv = f[((size_t)bb * 256 + d) * 128 + t];
        float dd = xv - fv;
        s += dd * dd;
    }
    red[tid] = s; __syncthreads();
    for (int s2 = 128; s2 > 0; s2 >>= 1) { if (tid < s2) red[tid] += red[tid + s2]; __syncthreads(); }
    if (tid == 0) atomicAdd(acc, red[0]);
}

__global__ void finalize_kernel(const float* __restrict__ acc, float* __restrict__ out)
{
    if (threadIdx.x == 0 && blockIdx.x == 0) {
        float t  = acc[0] / (128.f * 2048.f);
        float ar = acc[1] / (128.f * 128.f * 256.f);
        out[0] = 20.f * ar + 10.f * t;
        out[1] = t;
        out[2] = ar;
    }
}

// ---------------------------------------------------------------------------
// Host launcher
// ---------------------------------------------------------------------------
static void launch_gemm(hipStream_t s, const float* A, const float* Wt, const float* bias,
                        const float* res, float* C, int M, int N, int K, bool relu)
{
    dim3 grid(M / 64, N / 128), block(256);
    gemm_wmma_kernel<<<grid, block, 0, s>>>(A, Wt, bias, res, C, M, N, K, (int)relu);
}

extern "C" void kernel_launch(void* const* d_in, const int* in_sizes, int n_in,
                              void* d_out, int out_size, void* d_ws, size_t ws_size,
                              hipStream_t stream)
{
    (void)in_sizes; (void)n_in; (void)out_size; (void)ws_size;

    const float* xn  = (const float*)d_in[0];
    const float* ew[4]  = {(const float*)d_in[2],  (const float*)d_in[5],
                           (const float*)d_in[8],  (const float*)d_in[11]};
    const float* eg[4]  = {(const float*)d_in[3],  (const float*)d_in[6],
                           (const float*)d_in[9],  (const float*)d_in[12]};
    const float* eb[4]  = {(const float*)d_in[4],  (const float*)d_in[7],
                           (const float*)d_in[10], (const float*)d_in[13]};
    const float* dw[4]  = {(const float*)d_in[14], (const float*)d_in[17],
                           (const float*)d_in[20], (const float*)d_in[23]};
    const float* dg[3]  = {(const float*)d_in[15], (const float*)d_in[18], (const float*)d_in[21]};
    const float* db[3]  = {(const float*)d_in[16], (const float*)d_in[19], (const float*)d_in[22]};
    const float* cls  = (const float*)d_in[24];
    const float* wqkv = (const float*)d_in[25];
    const float* bqkv = (const float*)d_in[26];
    const float* wo   = (const float*)d_in[27];
    const float* bo   = (const float*)d_in[28];
    const float* ln1g = (const float*)d_in[29];
    const float* ln1b = (const float*)d_in[30];
    const float* ff1w = (const float*)d_in[31];
    const float* ff1b = (const float*)d_in[32];
    const float* ff2w = (const float*)d_in[33];
    const float* ff2b = (const float*)d_in[34];
    const float* ln2g = (const float*)d_in[35];
    const float* ln2b = (const float*)d_in[36];

    // workspace carve-up (floats)
    float* W = (float*)d_ws;
    size_t off = 0;
    auto alloc = [&](size_t n) { float* p = W + off; off += n; return p; };
    float* XC  = alloc((size_t)128 * 2048);
    float* A0  = alloc((size_t)8 * 1024 * 1024);
    float* A1  = alloc((size_t)8 * 1024 * 1024);
    float* ST  = alloc(1024);
    float* F   = alloc((size_t)128 * 256 * 128);
    float* X   = alloc((size_t)16512 * 256);
    float* XT  = alloc((size_t)16512 * 256);
    float* QK  = alloc((size_t)16512 * 768);
    float* CT  = alloc((size_t)16512 * 256);
    float* FFb = alloc((size_t)16512 * 512);
    float* RC  = alloc((size_t)128 * 2048);
    float* AC  = alloc(8);

    zero_kernel<<<1, 256, 0, stream>>>(AC, 8);

    // 1. wavelet denoise
    wavelet_kernel<<<128, 256, 0, stream>>>(xn, XC);

    // 2. encoder: WMMA conv -> bn stats -> bn+relu+maxpool, x4
    {
        const int cin[4]  = {1, 32, 64, 128};
        const int cout[4] = {32, 64, 128, 256};
        const float* cur = XC;
        int L = 2048;
        for (int l = 0; l < 4; ++l) {
            dim3 grid((128 * L) / 128, cout[l] / 32);
            conv_wmma_kernel<<<grid, 256, 0, stream>>>(cur, ew[l], A0, 128, cin[l], cout[l], L, L, 0);
            bn_stats_kernel<<<cout[l], 256, 0, stream>>>(A0, ST, 128, cout[l], L);
            float* dst = (l == 3) ? F : A1;
            int tot2 = 128 * cout[l] * (L / 2);
            bn_relu_pool_kernel<<<(tot2 + 255) / 256, 256, 0, stream>>>(A0, ST, eg[l], eb[l], dst, 128, cout[l], L);
            cur = dst;
            L >>= 1;
        }
    }

    // 3. decoder: WMMA repeat2+conv -> bn+relu (x3), final repeat2+conv (VALU)
    {
        const int dcin[4]  = {256, 128, 64, 32};
        const int dcout[4] = {128, 64, 32, 1};
        const float* cur = F;
        int Lin = 128;
        for (int l = 0; l < 3; ++l) {
            int Lo = 2 * Lin;
            dim3 grid((128 * Lo) / 128, dcout[l] / 32);
            conv_wmma_kernel<<<grid, 256, 0, stream>>>(cur, dw[l], A0, 128, dcin[l], dcout[l], Lin, Lo, 1);
            bn_stats_kernel<<<dcout[l], 256, 0, stream>>>(A0, ST, 128, dcout[l], Lo);
            int tot = 128 * dcout[l] * Lo;
            bn_relu_kernel<<<(tot + 255) / 256, 256, 0, stream>>>(A0, ST, dg[l], db[l], A1, dcout[l], Lo, tot);
            cur = A1;
            Lin = Lo;
        }
        int tot = 128 * 1 * 2048;
        convup_kernel<<<(tot + 255) / 256, 256, 0, stream>>>(cur, dw[3], RC, 128, 32, 1, 1024);
    }

    // t_loss_c accumulation
    sqdiff_kernel<<<512, 256, 0, stream>>>(RC, XC, AC, 128 * 2048);

    // 4. transformer (M = 128*129 = 16512)
    const int M = 16512;
    {
        int tot = 128 * 129 * 256;
        build_x0_kernel<<<(tot + 255) / 256, 256, 0, stream>>>(F, cls, X);
    }
    for (int l = 0; l < 4; ++l) {
        launch_gemm(stream, X, wqkv + (size_t)l * 768 * 256, bqkv + l * 768,
                    nullptr, QK, M, 768, 256, false);
        attention_kernel<<<512, 256, 0, stream>>>(QK, CT);
        launch_gemm(stream, CT, wo + (size_t)l * 256 * 256, bo + l * 256,
                    X, XT, M, 256, 256, false);
        layernorm_kernel<<<M, 256, 0, stream>>>(XT, ln1g + l * 256, ln1b + l * 256, X);
        launch_gemm(stream, X, ff1w + (size_t)l * 512 * 256, ff1b + l * 512,
                    nullptr, FFb, M, 512, 256, true);
        launch_gemm(stream, FFb, ff2w + (size_t)l * 256 * 512, ff2b + l * 256,
                    X, XT, M, 256, 512, false);
        layernorm_kernel<<<M, 256, 0, stream>>>(XT, ln2g + l * 256, ln2b + l * 256, X);
    }

    // ar_loss accumulation + finalize
    ar_loss_kernel<<<512, 256, 0, stream>>>(X, F, AC + 1);
    finalize_kernel<<<1, 1, 0, stream>>>(AC, (float*)d_out);
}
